// DeformableTransformerDecoderLayer_62929860821621
// MI455X (gfx1250) — matrix operand order repaired and checked by reading, not compiled, not run on hardware
//
#include <hip/hip_runtime.h>
#include <math.h>

#define D_MODEL 256
#define NHEAD   8
#define DHEAD   32
#define NQ      900
#define BS      8
#define TOT     13294
#define NQB     (NQ * BS)        /* 7200 */
#define MVAL    (TOT * BS)       /* 106352 */

typedef __attribute__((ext_vector_type(16))) __bf16 v16bf;
typedef __attribute__((ext_vector_type(8)))  float  v8f;

__device__ __forceinline__ unsigned short f2bf(float x) {
  unsigned u = __float_as_uint(x);
  u += 0x7FFFu + ((u >> 16) & 1u);          // round-to-nearest-even
  return (unsigned short)(u >> 16);
}
__device__ __forceinline__ float bf2f(unsigned short u) {
  return __uint_as_float(((unsigned)u) << 16);
}

__device__ __forceinline__ v8f wmma_bf16(v16bf a, v16bf b, v8f c) {
  return __builtin_amdgcn_wmma_f32_16x16x32_bf16(false, a, false, b, (short)0, c, false, false);
}

// Async global->LDS copy, 16B per lane, GVS mode (SGPR base + 32-bit VGPR offset).
__device__ __forceinline__ void async_b128(unsigned lds_addr, unsigned long long gbase, unsigned off) {
  asm volatile("global_load_async_to_lds_b128 %0, %1, %2"
               :: "v"(lds_addr), "v"(off), "s"(gbase) : "memory");
}
__device__ __forceinline__ void wait_async0() {
  asm volatile("s_wait_asynccnt 0x0" ::: "memory");
}
// ASYNCcnt completes in-order: <=3 outstanding means the previous tile's 3 copies landed.
__device__ __forceinline__ void wait_async3() {
  asm volatile("s_wait_asynccnt 0x3" ::: "memory");
}

// A fragment: 16(M) x 32(K) bf16 from LDS tile (row-major, stride in elems, even).
__device__ __forceinline__ v16bf load_a_frag(const unsigned short* lds, int row0, int stride) {
  int lane = threadIdx.x & 31;
  int m = lane & 15, hf = lane >> 4;
  union { v16bf v; unsigned u[8]; } f;
  const unsigned* p = (const unsigned*)lds;
#pragma unroll
  for (int i = 0; i < 8; ++i) {
    int kb = ((i >> 2) << 4) + (hf << 3) + ((i & 3) << 1);
    f.u[i] = p[((row0 + m) * stride + kb) >> 1];
  }
  return f.v;
}

// B fragment: 32(K) x 16(N) bf16; LDS tile stored as [n][k] row-major.
__device__ __forceinline__ v16bf load_b_frag(const unsigned short* lds, int n0, int stride) {
  int lane = threadIdx.x & 31;
  int n = lane & 15, hf = lane >> 4;
  union { v16bf v; unsigned u[8]; } f;
  const unsigned* p = (const unsigned*)lds;
#pragma unroll
  for (int i = 0; i < 8; ++i) {
    int kb = ((i >> 2) << 4) + (hf << 3) + ((i & 3) << 1);
    f.u[i] = p[((n0 + n) * stride + kb) >> 1];
  }
  return f.v;
}

// ------- GEMM: Y = X(MxK,bf16) @ W(NxK,bf16)^T + bias (+res)(+relu) ----------
#define BM 128
#define BN 64
#define LDT 40   // padded LDS row stride (bf16) -> 80B rows, 16B aligned chunks

template <bool RELU>
__global__ __launch_bounds__(256) void gemm_kernel(
    const unsigned short* __restrict__ X, const unsigned short* __restrict__ Wt,
    const float* __restrict__ bias, const float* __restrict__ res,
    float* __restrict__ Y, unsigned short* __restrict__ Ybf,
    int M, int N, int K) {
  __shared__ __align__(16) unsigned short Xs[2][BM * LDT];
  __shared__ __align__(16) unsigned short Wsm[2][BN * LDT];
  int t = threadIdx.x;
  int wave = t >> 5, lane = t & 31;
  int bm = blockIdx.y * BM, bn = blockIdx.x * BN;
  int wm = (wave >> 1) * 32, wn = (wave & 1) * 32;
  unsigned xs_base = (unsigned)(size_t)(void*)&Xs[0][0];
  unsigned wsm_base = (unsigned)(size_t)(void*)&Wsm[0][0];
  unsigned long long xg = (unsigned long long)(size_t)X;
  unsigned long long wg = (unsigned long long)(size_t)Wt;
  v8f acc00 = {}, acc01 = {}, acc10 = {}, acc11 = {};
  int ksteps = K >> 5;

  auto stage = [&](int ks, int buf) {
    int k0 = ks << 5;
    unsigned xb = xs_base + (unsigned)(buf * BM * LDT * 2);
    unsigned wb = wsm_base + (unsigned)(buf * BN * LDT * 2);
    int chunk = lane & 3;
#pragma unroll
    for (int s = 0; s < 2; ++s) {          // X tile: 128 rows x 64B, 8 rows/instr
      int row = wave * 16 + s * 8 + (lane >> 2);
      int gr = bm + row; if (gr > M - 1) gr = M - 1;
      unsigned off = (unsigned)(((size_t)gr * K + k0) * 2) + chunk * 16;
      async_b128(xb + row * (LDT * 2) + chunk * 16, xg, off);
    }
    int wr = wave * 8 + (lane >> 2);       // W tile: 64 rows x 64B
    int gw = bn + wr; if (gw > N - 1) gw = N - 1;
    unsigned off = (unsigned)(((size_t)gw * K + k0) * 2) + chunk * 16;
    async_b128(wb + wr * (LDT * 2) + chunk * 16, wg, off);
  };

  stage(0, 0);                             // prologue: tile 0 in flight
  for (int ks = 0; ks < ksteps; ++ks) {
    __syncthreads();                       // WAR: buffer (ks+1)&1 free of readers
    if (ks + 1 < ksteps) { stage(ks + 1, (ks + 1) & 1); wait_async3(); }
    else                 { wait_async0(); }
    __syncthreads();                       // all waves' tile-ks copies visible
    const unsigned short* xs = Xs[ks & 1];
    const unsigned short* wl = Wsm[ks & 1];
    v16bf a0 = load_a_frag(xs, wm, LDT);
    v16bf a1 = load_a_frag(xs, wm + 16, LDT);
    v16bf b0 = load_b_frag(wl, wn, LDT);
    v16bf b1 = load_b_frag(wl, wn + 16, LDT);
    acc00 = wmma_bf16(a0, b0, acc00);
    acc01 = wmma_bf16(a0, b1, acc01);
    acc10 = wmma_bf16(a1, b0, acc10);
    acc11 = wmma_bf16(a1, b1, acc11);
  }
  int n = lane & 15, hf = lane >> 4;
  auto store = [&](int row, int col, float v) {
    if (row < M && col < N) {
      float o = v + bias[col];
      if (res) o += res[(size_t)row * N + col];
      if (RELU) o = fmaxf(o, 0.f);
      if (Y)   Y[(size_t)row * N + col] = o;
      if (Ybf) Ybf[(size_t)row * N + col] = f2bf(o);
    }
  };
#pragma unroll
  for (int r = 0; r < 8; ++r) {
    int r0 = bm + wm + r + 8 * hf;
    int c0 = bn + wn + n;
    store(r0, c0, acc00[r]);
    store(r0, c0 + 16, acc01[r]);
    store(r0 + 16, c0, acc10[r]);
    store(r0 + 16, c0 + 16, acc11[r]);
  }
}

// ---------------- flash attention: per (b,h), 8 waves x 16 queries ------------
__global__ __launch_bounds__(256) void attn_kernel(
    const unsigned short* __restrict__ qp, const unsigned short* __restrict__ kp,
    const unsigned short* __restrict__ vp, unsigned short* __restrict__ outp) {
  __shared__ __align__(16) unsigned short Qs[8 * 16 * 34];
  __shared__ __align__(16) unsigned short Ks[32 * 34];
  __shared__ __align__(16) unsigned short Vt[32 * 34];
  __shared__ __align__(16) unsigned short Ps[8 * 16 * 34];
  int t = threadIdx.x;
  int wave = t >> 5, lane = t & 31;
  int b = blockIdx.y >> 3, h = blockIdx.y & 7;
  int q0 = blockIdx.x * 128 + wave * 16;
  unsigned short* Qw = Qs + wave * (16 * 34);
  unsigned short* Pw = Ps + wave * (16 * 34);
  {  // stage this wave's 16x32 Q tile (bf16 passthrough)
    int row = lane >> 1, cb = (lane & 1) * 16;
    int q = q0 + row;
    const unsigned short* src = qp + ((size_t)q * BS + b) * D_MODEL + h * DHEAD + cb;
#pragma unroll
    for (int j = 0; j < 16; ++j)
      Qw[row * 34 + cb + j] = (q < NQ) ? src[j] : (unsigned short)0;
  }
  v8f o0 = {}, o1 = {};
  float mrow[8], lrow[8];
#pragma unroll
  for (int r = 0; r < 8; ++r) { mrow[r] = -1e30f; lrow[r] = 0.f; }
  int n = lane & 15, hf = lane >> 4;
  const float scale = 0.17677669529663687f;  // 1/sqrt(32)
  for (int kb = 0; kb < NQ; kb += 32) {
    __syncthreads();
    {  // K tile: [key][d]
      int key = t >> 3, cb = (t & 7) * 4;
      int g = kb + key;
      const unsigned short* src = kp + ((size_t)g * BS + b) * D_MODEL + h * DHEAD + cb;
#pragma unroll
      for (int j = 0; j < 4; ++j)
        Ks[key * 34 + cb + j] = (g < NQ) ? src[j] : (unsigned short)0;
    }
    {  // V^T tile: [d][key]
      int d = t >> 3, c0 = (t & 7) * 4;
#pragma unroll
      for (int j = 0; j < 4; ++j) {
        int key = c0 + j, g = kb + key;
        Vt[d * 34 + key] = (g < NQ)
            ? vp[((size_t)g * BS + b) * D_MODEL + h * DHEAD + d] : (unsigned short)0;
      }
    }
    __syncthreads();
    v16bf qf = load_a_frag(Qw, 0, 34);
    v8f zero = {};
    v8f s0 = wmma_bf16(qf, load_b_frag(Ks, 0, 34), zero);
    v8f s1 = wmma_bf16(qf, load_b_frag(Ks, 16, 34), zero);
    bool ok0 = (kb + n) < NQ, ok1 = (kb + 16 + n) < NQ;
#pragma unroll
    for (int r = 0; r < 8; ++r) {
      float a = ok0 ? s0[r] * scale : -1e30f;
      float c = ok1 ? s1[r] * scale : -1e30f;
      float rm = fmaxf(a, c);
#pragma unroll
      for (int m = 1; m < 16; m <<= 1) rm = fmaxf(rm, __shfl_xor(rm, m, 16));
      float mnew = fmaxf(mrow[r], rm);
      float alpha = __expf(mrow[r] - mnew);
      mrow[r] = mnew;
      float p0 = __expf(a - mnew);
      float p1 = __expf(c - mnew);
      float rs = p0 + p1;
#pragma unroll
      for (int m = 1; m < 16; m <<= 1) rs += __shfl_xor(rs, m, 16);
      lrow[r] = lrow[r] * alpha + rs;
      o0[r] *= alpha; o1[r] *= alpha;
      Pw[(r + 8 * hf) * 34 + n] = f2bf(p0);          // C-layout -> A-layout
      Pw[(r + 8 * hf) * 34 + 16 + n] = f2bf(p1);     // via wave-private LDS
    }
    v16bf pf = load_a_frag(Pw, 0, 34);
    o0 = wmma_bf16(pf, load_b_frag(Vt, 0, 34), o0);
    o1 = wmma_bf16(pf, load_b_frag(Vt, 16, 34), o1);
  }
#pragma unroll
  for (int r = 0; r < 8; ++r) {
    int q = q0 + r + 8 * hf;
    if (q < NQ) {
      float inv = 1.f / lrow[r];
      size_t base = ((size_t)q * BS + b) * D_MODEL + h * DHEAD;
      outp[base + n] = f2bf(o0[r] * inv);
      outp[base + 16 + n] = f2bf(o1[r] * inv);
    }
  }
}

// -------- layernorm in place (+ optional bf16 copy); one wave per row --------
__global__ __launch_bounds__(256) void ln_kernel(float* __restrict__ x,
                                                 const float* __restrict__ g,
                                                 const float* __restrict__ bta,
                                                 unsigned short* __restrict__ ybf, int M) {
  int wave = threadIdx.x >> 5, lane = threadIdx.x & 31;
  int row = blockIdx.x * 8 + wave;
  if (row >= M) return;
  float v[8], s = 0.f, sq = 0.f;
  size_t base = (size_t)row * D_MODEL + lane;
#pragma unroll
  for (int j = 0; j < 8; ++j) { v[j] = x[base + j * 32]; s += v[j]; sq += v[j] * v[j]; }
#pragma unroll
  for (int m = 1; m < 32; m <<= 1) { s += __shfl_xor(s, m, 32); sq += __shfl_xor(sq, m, 32); }
  float mean = s * (1.f / D_MODEL);
  float var = sq * (1.f / D_MODEL) - mean * mean;
  float rstd = rsqrtf(var + 1e-5f);
#pragma unroll
  for (int j = 0; j < 8; ++j) {
    int col = lane + j * 32;
    float o = (v[j] - mean) * rstd * g[col] + bta[col];
    x[base + j * 32] = o;
    if (ybf) ybf[base + j * 32] = f2bf(o);
  }
}

__global__ __launch_bounds__(256) void add_bf16_kernel(const float* __restrict__ a,
                                                       const float* __restrict__ b,
                                                       unsigned short* __restrict__ y, int n) {
  int i = blockIdx.x * 256 + threadIdx.x;
  if (i < n) y[i] = f2bf(a[i] + b[i]);
}

__global__ __launch_bounds__(256) void cvt_bf16_kernel(const float* __restrict__ in,
                                                       unsigned short* __restrict__ outp, int n) {
  int i = blockIdx.x * 256 + threadIdx.x;
  if (i < n) outp[i] = f2bf(in[i]);
}

__global__ __launch_bounds__(256) void softmax16_kernel(const float* __restrict__ in,
                                                        float* __restrict__ outp) {
  int i = blockIdx.x * 256 + threadIdx.x;
  if (i >= NQB * NHEAD) return;
  const float* p = in + (size_t)i * 16;
  float mx = p[0];
#pragma unroll
  for (int j = 1; j < 16; ++j) mx = fmaxf(mx, p[j]);
  float e[16], s = 0.f;
#pragma unroll
  for (int j = 0; j < 16; ++j) { e[j] = __expf(p[j] - mx); s += e[j]; }
  float inv = 1.f / s;
  float* q = outp + (size_t)i * 16;
#pragma unroll
  for (int j = 0; j < 16; ++j) q[j] = e[j] * inv;
}

// ------------- multi-scale deformable sampling; one wave per (q,b,h) ----------
__global__ __launch_bounds__(256) void msdeform_kernel(
    const unsigned short* __restrict__ value, const float* __restrict__ off,
    const float* __restrict__ aw, const float* __restrict__ ref,
    unsigned short* __restrict__ outp) {
  int wave = threadIdx.x >> 5, lane = threadIdx.x & 31;
  int wid = blockIdx.x * 8 + wave;       // (q*BS + b)*NHEAD + h
  int h = wid & 7;
  int qb = wid >> 3;                     // q*BS + b
  int b = qb & 7;
  const int HL[4] = {100, 50, 25, 13};
  const int ST[4] = {0, 10000, 12500, 13125};
  float acc = 0.f;
#pragma unroll
  for (int l = 0; l < 4; ++l) {
    int Wl = HL[l], Hl = HL[l];
    float fW = (float)Wl, fH = (float)Hl;
#pragma unroll
    for (int p = 0; p < 4; ++p) {
      int oidx = qb * 256 + ((h * 4 + l) * 4 + p) * 2;
      float lx = ref[(qb * 4 + l) * 2 + 0] + off[oidx + 0] / fW;
      float ly = ref[(qb * 4 + l) * 2 + 1] + off[oidx + 1] / fH;
      float x = lx * fW - 0.5f, y = ly * fH - 0.5f;
      float wgt = aw[qb * 128 + (h * 4 + l) * 4 + p];
      float x0f = floorf(x), y0f = floorf(y);
      int x0 = (int)x0f, y0 = (int)y0f;
      float wx1 = x - x0f, wx0 = 1.f - wx1;
      float wy1 = y - y0f, wy0 = 1.f - wy1;
#pragma unroll
      for (int c = 0; c < 4; ++c) {
        int xi = x0 + (c & 1), yi = y0 + (c >> 1);
        float cw = (c & 1 ? wx1 : wx0) * (c >> 1 ? wy1 : wy0);
        if (xi >= 0 && xi < Wl && yi >= 0 && yi < Hl) {
          float v = bf2f(value[((size_t)(ST[l] + yi * Wl + xi) * BS + b) * D_MODEL + h * DHEAD + lane]);
          acc += wgt * cw * v;
        }
      }
    }
  }
  outp[(size_t)qb * D_MODEL + h * DHEAD + lane] = f2bf(acc);
}

// -----------------------------------------------------------------------------
extern "C" void kernel_launch(void* const* d_in, const int* in_sizes, int n_in,
                              void* d_out, int out_size, void* d_ws, size_t ws_size,
                              hipStream_t stream) {
  (void)in_sizes; (void)n_in; (void)out_size; (void)ws_size;
  const float* tgt = (const float*)d_in[0];
  const float* pos = (const float*)d_in[1];
  const float* ref = (const float*)d_in[2];
  const float* mem = (const float*)d_in[3];
  const float* sa_in_w  = (const float*)d_in[6];
  const float* sa_in_b  = (const float*)d_in[7];
  const float* sa_out_w = (const float*)d_in[8];
  const float* sa_out_b = (const float*)d_in[9];
  const float* n1g = (const float*)d_in[10]; const float* n1b = (const float*)d_in[11];
  const float* n2g = (const float*)d_in[12]; const float* n2b = (const float*)d_in[13];
  const float* n3g = (const float*)d_in[14]; const float* n3b = (const float*)d_in[15];
  const float* samp_w = (const float*)d_in[16]; const float* samp_b = (const float*)d_in[17];
  const float* attw_w = (const float*)d_in[18]; const float* attw_b = (const float*)d_in[19];
  const float* val_w  = (const float*)d_in[20]; const float* val_b  = (const float*)d_in[21];
  const float* outp_w = (const float*)d_in[22]; const float* outp_b = (const float*)d_in[23];
  const float* l1w = (const float*)d_in[24]; const float* l1b = (const float*)d_in[25];
  const float* l2w = (const float*)d_in[26]; const float* l2b = (const float*)d_in[27];
  float* out = (float*)d_out;

  const size_t NE = (size_t)NQB * D_MODEL;
  float* wsf  = (float*)d_ws;
  float* tgt1 = wsf;                   // s1, LN2 in place
  float* tgt2 = tgt1 + NE;             // s2, LN1 in place
  float* offb = tgt2 + NE;
  float* awl  = offb + NE;
  float* aw   = awl + (size_t)NQB * 128;
  unsigned short* q_bufb = (unsigned short*)(aw + (size_t)NQB * 128);
  unsigned short* tgtb   = q_bufb + NE;
  unsigned short* qpb    = tgtb + NE;
  unsigned short* kpb    = qpb + NE;
  unsigned short* vpb    = kpb + NE;
  unsigned short* attnb  = vpb + NE;
  unsigned short* q2b    = attnb + NE;
  unsigned short* msoutb = q2b + NE;
  unsigned short* tgt2b  = msoutb + NE;
  unsigned short* hidb   = tgt2b + NE;
  unsigned short* memb   = hidb + (size_t)NQB * 1024;
  unsigned short* valueb = memb + (size_t)MVAL * D_MODEL;
  unsigned short* w_in   = valueb + (size_t)MVAL * D_MODEL;
  unsigned short* w_out  = w_in  + 768 * 256;
  unsigned short* w_samp = w_out + 256 * 256;
  unsigned short* w_attw = w_samp + 256 * 256;
  unsigned short* w_val  = w_attw + 128 * 256;
  unsigned short* w_outp = w_val + 256 * 256;
  unsigned short* w_l1   = w_outp + 256 * 256;
  unsigned short* w_l2   = w_l1  + 1024 * 256;

  dim3 blk(256);
  dim3 g256(4, (NQB + BM - 1) / BM);
  auto cvt = [&](const float* s, unsigned short* d, size_t n) {
    cvt_bf16_kernel<<<(unsigned)((n + 255) / 256), blk, 0, stream>>>(s, d, (int)n);
  };

  // one-time bf16 conversions (weights + memory + tgt)
  cvt(mem, memb, (size_t)MVAL * D_MODEL);
  cvt(tgt, tgtb, NE);
  cvt(sa_in_w, w_in, 768 * 256);
  cvt(sa_out_w, w_out, 256 * 256);
  cvt(samp_w, w_samp, 256 * 256);
  cvt(attw_w, w_attw, 128 * 256);
  cvt(val_w, w_val, 256 * 256);
  cvt(outp_w, w_outp, 256 * 256);
  cvt(l1w, w_l1, 1024 * 256);
  cvt(l2w, w_l2, 256 * 1024);

  // q = tgt + pos (bf16)
  add_bf16_kernel<<<(int)(NE / 256), blk, 0, stream>>>(tgt, pos, q_bufb, (int)NE);
  // qkv projections (bf16 outputs)
  gemm_kernel<false><<<g256, blk, 0, stream>>>(q_bufb, w_in,             sa_in_b,       nullptr, nullptr, qpb, NQB, 256, 256);
  gemm_kernel<false><<<g256, blk, 0, stream>>>(q_bufb, w_in + 256 * 256, sa_in_b + 256, nullptr, nullptr, kpb, NQB, 256, 256);
  gemm_kernel<false><<<g256, blk, 0, stream>>>(tgtb,   w_in + 512 * 256, sa_in_b + 512, nullptr, nullptr, vpb, NQB, 256, 256);
  // self attention (flash, bf16 out)
  attn_kernel<<<dim3(8, 64), blk, 0, stream>>>(qpb, kpb, vpb, attnb);
  // out-proj + residual -> fp32, LN2 in place
  gemm_kernel<false><<<g256, blk, 0, stream>>>(attnb, w_out, sa_out_b, tgt, tgt1, nullptr, NQB, 256, 256);
  ln_kernel<<<NQB / 8, blk, 0, stream>>>(tgt1, n2g, n2b, nullptr, NQB);
  // q2 = tgt1 + pos (bf16)
  add_bf16_kernel<<<(int)(NE / 256), blk, 0, stream>>>(tgt1, pos, q2b, (int)NE);
  // sampling offsets / attention weights (fp32 outputs)
  gemm_kernel<false><<<g256, blk, 0, stream>>>(q2b, w_samp, samp_b, nullptr, offb, nullptr, NQB, 256, 256);
  gemm_kernel<false><<<dim3(2, (NQB + BM - 1) / BM), blk, 0, stream>>>(q2b, w_attw, attw_b, nullptr, awl, nullptr, NQB, 128, 256);
  softmax16_kernel<<<(NQB * NHEAD) / 256, blk, 0, stream>>>(awl, aw);
  // value projection over the whole memory (bf16 output for cheap gathers)
  gemm_kernel<false><<<dim3(4, (MVAL + BM - 1) / BM), blk, 0, stream>>>(memb, w_val, val_b, nullptr, nullptr, valueb, MVAL, 256, 256);
  // deformable sampling (bf16 out)
  msdeform_kernel<<<NQB, blk, 0, stream>>>(valueb, offb, aw, ref, msoutb);
  // output proj + residual -> fp32, LN1 in place (+ bf16 copy for FFN input)
  gemm_kernel<false><<<g256, blk, 0, stream>>>(msoutb, w_outp, outp_b, tgt1, tgt2, nullptr, NQB, 256, 256);
  ln_kernel<<<NQB / 8, blk, 0, stream>>>(tgt2, n1g, n1b, tgt2b, NQB);
  // FFN
  gemm_kernel<true ><<<dim3(16, (NQB + BM - 1) / BM), blk, 0, stream>>>(tgt2b, w_l1, l1b, nullptr, nullptr, hidb, NQB, 1024, 256);
  gemm_kernel<false><<<g256, blk, 0, stream>>>(hidb, w_l2, l2b, tgt2, out, nullptr, NQB, 256, 1024);
  ln_kernel<<<NQB / 8, blk, 0, stream>>>(out, n3g, n3b, nullptr, NQB);
}
